// EfficientAttention_39556648796548
// MI455X (gfx1250) — matrix operand order repaired and checked by reading, not compiled, not run on hardware
//
#include <hip/hip_runtime.h>
#include <hip/hip_bf16.h>

typedef __attribute__((ext_vector_type(16))) __bf16 v16bf;
typedef __attribute__((ext_vector_type(8)))  __bf16 v8bf;
typedef __attribute__((ext_vector_type(8)))  float  v8f;

#define WMMA_BF16(a, b, c) \
  __builtin_amdgcn_wmma_f32_16x16x32_bf16(false, (a), false, (b), (short)0, (c), false, false)

// ---------------------------------------------------------------------------
// Tile loaders for gfx1250 WMMA operands in contiguous k order (A and B alike, not the ISA table order).
// A (16x32 bf16, MxK): lane = M (lane%16); lanes 0-15 hold K0-15 in v0-7,
// lanes 16-31 hold K16-31.  -> one contiguous 32B run (two 16B loads)/lane.
// B (32x16 bf16, KxN): lane = N (lane%16); lanes 0-15 hold K0-15 in v0-7,
// lanes 16-31 hold K16-31.  -> one contiguous 32B run (two 16B loads)/lane.
// ---------------------------------------------------------------------------
static __device__ __forceinline__ v16bf load_a_g(const __bf16* p, int row, int ld, int kbase) {
  const int lane = threadIdx.x & 31;
  const int r = lane & 15;
  const int ko = (lane >> 4) * 16;
  const __bf16* q = p + (size_t)(row + r) * ld + kbase + ko;
  v8bf lo = *(const v8bf*)q;
  v8bf hi = *(const v8bf*)(q + 8);
  v16bf out;
#pragma unroll
  for (int i = 0; i < 8; ++i) { out[i] = lo[i]; out[i + 8] = hi[i]; }
  return out;
}

static __device__ __forceinline__ v16bf load_b_g(const __bf16* p, int col, int ld, int kbase) {
  const int lane = threadIdx.x & 31;
  const int n = lane & 15;
  const int ko = (lane >> 4) * 16;
  const __bf16* q = p + (size_t)(col + n) * ld + kbase + ko;
  v8bf lo = *(const v8bf*)q;
  v8bf hi = *(const v8bf*)(q + 8);
  v16bf out;
#pragma unroll
  for (int i = 0; i < 8; ++i) { out[i] = lo[i]; out[i + 8] = hi[i]; }
  return out;
}

// B fragment from the LDS-staged weight panel. Panel layout: 64 cols x 32 k,
// row stride padded to 40 elements (80B) so the 16 lanes of a half hit 16
// distinct banks (n*20 mod 64 is a permutation for n in [0,16)).
static __device__ __forceinline__ v16bf load_b_lds(const __bf16* p, int col) {
  const int lane = threadIdx.x & 31;
  const int n = lane & 15;
  const int ko = (lane >> 4) * 16;
  const __bf16* q = p + (col + n) * 40 + ko;
  v8bf lo = *(const v8bf*)q;
  v8bf hi = *(const v8bf*)(q + 8);
  v16bf out;
#pragma unroll
  for (int i = 0; i < 8; ++i) { out[i] = lo[i]; out[i + 8] = hi[i]; }
  return out;
}

// ---------------------------------------------------------------------------
// Async global->LDS staging of a 64col x 32k bf16 weight panel (4 KiB).
// 128 threads x 2 chunks x 16B; each wave issues 2 async instructions
// (tracked on its own ASYNCcnt; in-order completion).
// ---------------------------------------------------------------------------
static __device__ __forceinline__ void async_copy_bpanel(const __bf16* Wsrc, int bcolBase,
                                                         int k, __bf16* dstShared) {
  const int t = threadIdx.x;  // 0..127
  const unsigned lbase = (unsigned)(unsigned long long)(uintptr_t)dstShared;
#pragma unroll
  for (int c = 0; c < 2; ++c) {
    const int chunk = t + c * 128;       // 0..255
    const int row = chunk >> 2;          // 0..63 (output column)
    const int off = (chunk & 3) * 8;     // bf16 elems; 16B granules within a row
    const unsigned long long gaddr =
        (unsigned long long)(uintptr_t)(Wsrc + (size_t)(bcolBase + row) * 1024 + k + off);
    const unsigned laddr = lbase + (unsigned)(row * 40 + off) * 2;
    asm volatile("global_load_async_to_lds_b128 %0, %1, off"
                 :: "v"(laddr), "v"(gaddr) : "memory");
  }
}

#define WAIT_ASYNCCNT(n) asm volatile("s_wait_asynccnt " #n ::: "memory")

// ---------------------------------------------------------------------------
// fp32 -> bf16 conversion (one-time, bandwidth-trivial vs. the GEMMs)
// ---------------------------------------------------------------------------
__global__ void cvt_f32_bf16(const float* __restrict__ src, __bf16* __restrict__ dst, int n) {
  int i = blockIdx.x * blockDim.x + threadIdx.x;
  int stride = gridDim.x * blockDim.x;
  for (; i < n; i += stride) dst[i] = (__bf16)src[i];
}

// ---------------------------------------------------------------------------
// Fused QKV projection: [4096,1024] @ W^T + b, plus sinusoidal PE for Q/K,
// transposed store for V.  Block = 128 threads (4 waves), block tile 64x64,
// wave tile 32x32.  Weight panel double-buffered through LDS via async DMA.
// ---------------------------------------------------------------------------
__global__ __launch_bounds__(128)
void qkv_proj(const __bf16* __restrict__ xq, const __bf16* __restrict__ xk,
              const __bf16* __restrict__ xv,
              const __bf16* __restrict__ wq, const __bf16* __restrict__ wk,
              const __bf16* __restrict__ wv,
              const float* __restrict__ bq, const float* __restrict__ bk,
              const float* __restrict__ bv,
              __bf16* __restrict__ qh, __bf16* __restrict__ kh,
              __bf16* __restrict__ vhT) {
  __shared__ __bf16 bpanel[2][64 * 40];
  const int which = blockIdx.z;
  const __bf16* X = (which == 0) ? xq : (which == 1) ? xk : xv;
  const __bf16* W = (which == 0) ? wq : (which == 1) ? wk : wv;
  const float* bias = (which == 0) ? bq : (which == 1) ? bk : bv;

  const int wave = threadIdx.x >> 5;
  const int lane = threadIdx.x & 31;
  const int arow = blockIdx.x * 64 + (wave >> 1) * 32;
  const int bcolBase = blockIdx.y * 64;
  const int bcolLoc = (wave & 1) * 32;

  async_copy_bpanel(W, bcolBase, 0, &bpanel[0][0]);

  v8f c[2][2] = {};
  for (int k = 0; k < 1024; k += 32) {
    const int buf = (k >> 5) & 1;
    __syncthreads();  // everyone done reading bpanel[buf^1] from the previous iter
    if (k + 32 < 1024) {
      async_copy_bpanel(W, bcolBase, k + 32, &bpanel[buf ^ 1][0]);
      WAIT_ASYNCCNT(0x2);  // retire batch k (in-order), keep batch k+32 in flight
    } else {
      WAIT_ASYNCCNT(0x0);
    }
    __syncthreads();  // panel k visible to all waves

    v16bf a0 = load_a_g(X, arow,      1024, k);
    v16bf a1 = load_a_g(X, arow + 16, 1024, k);
    v16bf b0 = load_b_lds(&bpanel[buf][0], bcolLoc);
    v16bf b1 = load_b_lds(&bpanel[buf][0], bcolLoc + 16);
    c[0][0] = WMMA_BF16(a0, b0, c[0][0]);
    c[0][1] = WMMA_BF16(a0, b1, c[0][1]);
    c[1][0] = WMMA_BF16(a1, b0, c[1][0]);
    c[1][1] = WMMA_BF16(a1, b1, c[1][1]);
  }

  const int half = lane >> 4, nl = lane & 15;
#pragma unroll
  for (int ti = 0; ti < 2; ++ti)
#pragma unroll
    for (int tj = 0; tj < 2; ++tj)
#pragma unroll
      for (int i = 0; i < 8; ++i) {
        const int m = arow + ti * 16 + i + 8 * half;       // global row in [B*S]
        const int n = bcolBase + bcolLoc + tj * 16 + nl;   // global col in [D]
        float v = c[ti][tj][i] + bias[n];
        const int b = m >> 10, s = m & 1023, h = n >> 6, d = n & 63;
        const size_t bh = (size_t)b * 16 + h;
        if (which < 2) {
          // sinusoidal PE: pe[s, d] = sin/cos(s * 10000^(-(d&~1)/64))
          const float ang = (float)s * __expf((float)(d & ~1) * -0.14391157f);
          v += (d & 1) ? __cosf(ang) : __sinf(ang);
          __bf16* dst = (which == 0) ? qh : kh;
          dst[(bh * 1024 + s) * 64 + d] = (__bf16)v;       // [B,H,S,hd]
        } else {
          vhT[(bh * 64 + d) * 1024 + s] = (__bf16)v;       // [B,H,hd,S]
        }
      }
}

// ---------------------------------------------------------------------------
// Flash attention: grid (S/64, H, B), 4 waves/block, each wave owns 16 query
// rows with online softmax.  Per 32-key block: 4 WMMA (scores, K=64 chained)
// + 4 WMMA (P@V over 4 d-chunks).  P is re-layouted C->A through LDS
// (per-wave region; same-wave DS ops are in-order on CDNA5).
// ---------------------------------------------------------------------------
__global__ __launch_bounds__(128)
void flash_attn(const __bf16* __restrict__ qh, const __bf16* __restrict__ kh,
                const __bf16* __restrict__ vhT, __bf16* __restrict__ attOut) {
  __shared__ __bf16 pshm[4 * 16 * 32];
  const int wave = threadIdx.x >> 5, lane = threadIdx.x & 31;
  const int half = lane >> 4, nl = lane & 15;
  const int h = blockIdx.y, b = blockIdx.z;
  const size_t bh = (size_t)b * 16 + h;
  const __bf16* Q = qh + bh * 1024 * 64;
  const __bf16* K = kh + bh * 1024 * 64;
  const __bf16* V = vhT + bh * 64 * 1024;
  const int mrow = blockIdx.x * 64 + wave * 16;

  // Q tile stays resident: 16x64 as two A operands
  v16bf qa0 = load_a_g(Q, mrow, 64, 0);
  v16bf qa1 = load_a_g(Q, mrow, 64, 32);

  v8f o[4] = {};
  float mrun[8], lrun[8];
#pragma unroll
  for (int i = 0; i < 8; ++i) { mrun[i] = -__builtin_inff(); lrun[i] = 0.f; }

  __bf16* pl = &pshm[wave * 16 * 32];

  for (int kb = 0; kb < 1024; kb += 32) {
    v16bf kb0a = load_b_g(K, kb,      64, 0);
    v16bf kb0b = load_b_g(K, kb,      64, 32);
    v16bf kb1a = load_b_g(K, kb + 16, 64, 0);
    v16bf kb1b = load_b_g(K, kb + 16, 64, 32);
    v8f s0 = {}, s1 = {};
    s0 = WMMA_BF16(qa0, kb0a, s0);
    s0 = WMMA_BF16(qa1, kb0b, s0);
    s1 = WMMA_BF16(qa0, kb1a, s1);
    s1 = WMMA_BF16(qa1, kb1b, s1);

#pragma unroll
    for (int i = 0; i < 8; ++i) {
      float a0 = s0[i] * 0.125f;   // hd^-0.5
      float a1 = s1[i] * 0.125f;
      float mx = fmaxf(a0, a1);
#pragma unroll
      for (int off = 1; off < 16; off <<= 1) mx = fmaxf(mx, __shfl_xor(mx, off, 32));
      const float mnew = fmaxf(mrun[i], mx);
      const float alpha = __expf(mrun[i] - mnew);
      const float p0 = __expf(a0 - mnew);
      const float p1 = __expf(a1 - mnew);
      float rs = p0 + p1;
#pragma unroll
      for (int off = 1; off < 16; off <<= 1) rs += __shfl_xor(rs, off, 32);
      lrun[i] = lrun[i] * alpha + rs;
      mrun[i] = mnew;
#pragma unroll
      for (int j = 0; j < 4; ++j) o[j][i] *= alpha;
      const int row = i + 8 * half;                 // C layout -> row-major LDS
      pl[row * 32 + nl]      = (__bf16)p0;
      pl[row * 32 + 16 + nl] = (__bf16)p1;
    }

    // Reload P in A layout from LDS (ds_load), then O += P @ V
    {
      const int r = lane & 15, ko = half * 16;
      const __bf16* q = pl + r * 32 + ko;
      v8bf lo = *(const v8bf*)q;
      v8bf hi = *(const v8bf*)(q + 8);
      v16bf pa;
#pragma unroll
      for (int t = 0; t < 8; ++t) { pa[t] = lo[t]; pa[t + 8] = hi[t]; }
#pragma unroll
      for (int j = 0; j < 4; ++j) {
        v16bf vb = load_b_g(V, j * 16, 1024, kb);   // V^T rows are d, contraction = s
        o[j] = WMMA_BF16(pa, vb, o[j]);
      }
    }
  }

  // normalize and store into [B, S, H*hd] bf16 for the output projection
#pragma unroll
  for (int i = 0; i < 8; ++i) {
    const float inv = 1.f / lrun[i];
    const int s = mrow + i + 8 * half;
#pragma unroll
    for (int j = 0; j < 4; ++j) {
      const float v = o[j][i] * inv;
      attOut[((size_t)b * 1024 + s) * 1024 + h * 64 + j * 16 + nl] = (__bf16)v;
    }
  }
}

// ---------------------------------------------------------------------------
// Output projection: attOut [4096,1024] @ Wo^T + bo -> fp32 d_out
// Same async-LDS weight staging as qkv_proj.
// ---------------------------------------------------------------------------
__global__ __launch_bounds__(128)
void out_proj(const __bf16* __restrict__ A, const __bf16* __restrict__ W,
              const float* __restrict__ bias, float* __restrict__ out) {
  __shared__ __bf16 bpanel[2][64 * 40];
  const int wave = threadIdx.x >> 5;
  const int lane = threadIdx.x & 31;
  const int arow = blockIdx.x * 64 + (wave >> 1) * 32;
  const int bcolBase = blockIdx.y * 64;
  const int bcolLoc = (wave & 1) * 32;

  async_copy_bpanel(W, bcolBase, 0, &bpanel[0][0]);

  v8f c[2][2] = {};
  for (int k = 0; k < 1024; k += 32) {
    const int buf = (k >> 5) & 1;
    __syncthreads();
    if (k + 32 < 1024) {
      async_copy_bpanel(W, bcolBase, k + 32, &bpanel[buf ^ 1][0]);
      WAIT_ASYNCCNT(0x2);
    } else {
      WAIT_ASYNCCNT(0x0);
    }
    __syncthreads();

    v16bf a0 = load_a_g(A, arow,      1024, k);
    v16bf a1 = load_a_g(A, arow + 16, 1024, k);
    v16bf b0 = load_b_lds(&bpanel[buf][0], bcolLoc);
    v16bf b1 = load_b_lds(&bpanel[buf][0], bcolLoc + 16);
    c[0][0] = WMMA_BF16(a0, b0, c[0][0]);
    c[0][1] = WMMA_BF16(a0, b1, c[0][1]);
    c[1][0] = WMMA_BF16(a1, b0, c[1][0]);
    c[1][1] = WMMA_BF16(a1, b1, c[1][1]);
  }

  const int half = lane >> 4, nl = lane & 15;
#pragma unroll
  for (int ti = 0; ti < 2; ++ti)
#pragma unroll
    for (int tj = 0; tj < 2; ++tj)
#pragma unroll
      for (int i = 0; i < 8; ++i) {
        const int m = arow + ti * 16 + i + 8 * half;
        const int n = bcolBase + bcolLoc + tj * 16 + nl;
        out[(size_t)m * 1024 + n] = c[ti][tj][i] + bias[n];
      }
}

// ---------------------------------------------------------------------------
extern "C" void kernel_launch(void* const* d_in, const int* in_sizes, int n_in,
                              void* d_out, int out_size, void* d_ws, size_t ws_size,
                              hipStream_t stream) {
  (void)in_sizes; (void)n_in; (void)out_size; (void)ws_size;
  const float* q  = (const float*)d_in[0];
  const float* k  = (const float*)d_in[1];
  const float* v  = (const float*)d_in[2];
  const float* Wq = (const float*)d_in[3];
  const float* bq = (const float*)d_in[4];
  const float* Wk = (const float*)d_in[5];
  const float* bk = (const float*)d_in[6];
  const float* Wv = (const float*)d_in[7];
  const float* bv = (const float*)d_in[8];
  const float* Wo = (const float*)d_in[9];
  const float* bo = (const float*)d_in[10];
  float* out = (float*)d_out;

  char* ws = (char*)d_ws;
  const size_t ACT = (size_t)4096 * 1024 * 2;   // 8 MiB bf16 activation
  const size_t WGT = (size_t)1024 * 1024 * 2;   // 2 MiB bf16 weight
  __bf16* qb   = (__bf16*)(ws + 0 * ACT);
  __bf16* kb   = (__bf16*)(ws + 1 * ACT);
  __bf16* vb   = (__bf16*)(ws + 2 * ACT);
  __bf16* wqb  = (__bf16*)(ws + 3 * ACT + 0 * WGT);
  __bf16* wkb  = (__bf16*)(ws + 3 * ACT + 1 * WGT);
  __bf16* wvb  = (__bf16*)(ws + 3 * ACT + 2 * WGT);
  __bf16* wob  = (__bf16*)(ws + 3 * ACT + 3 * WGT);
  __bf16* qhp  = (__bf16*)(ws + 4 * ACT + 0 * WGT);
  __bf16* khp  = (__bf16*)(ws + 5 * ACT + 0 * WGT);
  __bf16* vhT  = (__bf16*)(ws + 6 * ACT + 0 * WGT);
  __bf16* attO = (__bf16*)(ws + 7 * ACT + 0 * WGT);

  const int NA = 4096 * 1024, NW = 1024 * 1024;
  cvt_f32_bf16<<<1024, 256, 0, stream>>>(q,  qb,  NA);
  cvt_f32_bf16<<<1024, 256, 0, stream>>>(k,  kb,  NA);
  cvt_f32_bf16<<<1024, 256, 0, stream>>>(v,  vb,  NA);
  cvt_f32_bf16<<<512,  256, 0, stream>>>(Wq, wqb, NW);
  cvt_f32_bf16<<<512,  256, 0, stream>>>(Wk, wkb, NW);
  cvt_f32_bf16<<<512,  256, 0, stream>>>(Wv, wvb, NW);
  cvt_f32_bf16<<<512,  256, 0, stream>>>(Wo, wob, NW);

  qkv_proj<<<dim3(64, 16, 3), 128, 0, stream>>>(qb, kb, vb, wqb, wkb, wvb,
                                                bq, bk, bv, qhp, khp, vhT);
  flash_attn<<<dim3(16, 16, 4), 128, 0, stream>>>(qhp, khp, vhT, attO);
  out_proj<<<dim3(64, 16, 1), 128, 0, stream>>>(attO, wob, bo, out);
}
